// SurfaceEncoding_69741678952565
// MI455X (gfx1250) — matrix operand order repaired and checked by reading, not compiled, not run on hardware
//
#include <hip/hip_runtime.h>

typedef __bf16 bf16;
typedef __attribute__((ext_vector_type(16))) __bf16 v16bf;
typedef __attribute__((ext_vector_type(8)))  float  v8f;

#define B_     32
#define C_     256
#define K_     64
#define N_     4096
#define NCHUNK 512
#define NCHUNKS_PER_B 8
#define NTILE  128
#define NT_PER_CHUNK (NCHUNK / NTILE)   // 4

// dynamic LDS partition (bytes)
#define OFF_X      0         // bf16 [C_][NTILE]     65536
#define OFF_CLUS   65536     // bf16 [K_][C_]        32768
#define OFF_S      98304     // f32  [K_][NTILE]     32768
#define OFF_THETA  131072    // bf16 [K_][NTILE]     16384
#define OFF_PART   147456    // f32  [8][NTILE]       4096
#define OFF_CSQ    151552    // f32  [K_]              256
#define OFF_SCALE  151808    // f32  [K_]              256
#define OFF_TSUM   152064    // f32  [K_]              256
#define OFF_XSQ    152320    // f32  [NTILE]           512
#define SMEM_BYTES 152832

union BfPair { unsigned int u; bf16 h[2]; };

__device__ __forceinline__ void ld_bf16x2(v16bf& v, int i, const bf16* p) {
  BfPair c; c.u = *(const unsigned int*)p;   // single ds_load_b32
  v[i] = c.h[0]; v[i + 1] = c.h[1];
}

__global__ __launch_bounds__(256)
void surface_encoding_fused(const float* __restrict__ x,
                            const float* __restrict__ clusters,
                            const float* __restrict__ scale_g,
                            float* __restrict__ out) {
  extern __shared__ char smem[];
  bf16*  sX     = (bf16*)(smem + OFF_X);
  bf16*  sClus  = (bf16*)(smem + OFF_CLUS);
  float* sS     = (float*)(smem + OFF_S);
  bf16*  sTheta = (bf16*)(smem + OFF_THETA);
  float* sPart  = (float*)(smem + OFF_PART);
  float* sCsq   = (float*)(smem + OFF_CSQ);
  float* sScale = (float*)(smem + OFF_SCALE);
  float* sTsum  = (float*)(smem + OFF_TSUM);
  float* sXsq   = (float*)(smem + OFF_XSQ);

  const int tid   = threadIdx.x;
  const int wave  = tid >> 5;
  const int lane  = tid & 31;
  const int g     = lane >> 4;      // half-wave group
  const int m     = lane & 15;      // position within group
  const int b     = blockIdx.x >> 3;
  const int chunk = blockIdx.x & 7;

  const float* xb   = x + (size_t)b * C_ * N_ + (size_t)chunk * NCHUNK;
  float*       outb = out + (size_t)b * K_ * C_;

  // ---- preload clusters (bf16), c_sq, scale ----
  for (int i = tid; i < (K_ * C_) / 4; i += 256) {
    float4 f = ((const float4*)clusters)[i];
    int base = i * 4;
    sClus[base + 0] = (bf16)f.x; sClus[base + 1] = (bf16)f.y;
    sClus[base + 2] = (bf16)f.z; sClus[base + 3] = (bf16)f.w;
  }
  if (tid < K_) {
    float s = 0.f;
    for (int c = 0; c < C_; ++c) { float v = clusters[tid * C_ + c]; s += v * v; }
    sCsq[tid]   = s;
    sScale[tid] = scale_g[tid];
    sTsum[tid]  = 0.f;
  }
  __syncthreads();

  // persistent E accumulators: 8 tiles of 16x16 f32 per wave  (E is C_ x K_)
  v8f e[8];
  const v8f vzero = {0.f, 0.f, 0.f, 0.f, 0.f, 0.f, 0.f, 0.f};
#pragma unroll
  for (int j = 0; j < 8; ++j) e[j] = vzero;

  const int kt  = wave & 3;     // 16-wide tile along K
  const int par = wave >> 2;    // tile parity along the other dim

  for (int t = 0; t < NT_PER_CHUNK; ++t) {
    const float* xt = xb + t * NTILE;

    // ---- stage x tile (f32 -> bf16) + partial x_sq in f32 ----
    {
      float p0 = 0.f, p1 = 0.f, p2 = 0.f, p3 = 0.f;
      for (int i = tid; i < (C_ * NTILE) / 4; i += 256) {
        int c  = i >> 5;       // NTILE/4 == 32 float4 per row
        int n4 = i & 31;
        float4 f = *(const float4*)(xt + (size_t)c * N_ + n4 * 4);
        int base = c * NTILE + n4 * 4;
        sX[base + 0] = (bf16)f.x; sX[base + 1] = (bf16)f.y;
        sX[base + 2] = (bf16)f.z; sX[base + 3] = (bf16)f.w;
        p0 += f.x * f.x; p1 += f.y * f.y; p2 += f.z * f.z; p3 += f.w * f.w;
      }
      int grp = tid >> 5, n4 = tid & 31;
      sPart[grp * NTILE + n4 * 4 + 0] = p0;
      sPart[grp * NTILE + n4 * 4 + 1] = p1;
      sPart[grp * NTILE + n4 * 4 + 2] = p2;
      sPart[grp * NTILE + n4 * 4 + 3] = p3;
    }

    // ---- prefetch next tile into cache while this tile computes ----
    // tile = 128 KB = 1024 x 128B cachelines; 4 lines per thread
    if (t + 1 < NT_PER_CHUNK) {
      const float* xn = xb + (t + 1) * NTILE;
#pragma unroll
      for (int l4 = 0; l4 < 4; ++l4) {
        int l = tid * 4 + l4;                    // line index 0..1023
        __builtin_prefetch(xn + (size_t)(l >> 2) * N_ + (l & 3) * 32, 0, 3);
      }
    }
    __syncthreads();

    if (tid < NTILE) {
      float s = 0.f;
#pragma unroll
      for (int q = 0; q < 8; ++q) s += sPart[q * NTILE + tid];
      sXsq[tid] = s;
    }

    // ---- phase 1: S(K_ x NTILE) = clusters(K_ x C_) * xtile(C_ x NTILE) ----
    {
      v8f acc[4];
#pragma unroll
      for (int j = 0; j < 4; ++j) acc[j] = vzero;

      for (int cb = 0; cb < C_; cb += 32) {
        // A fragment: clusters row k = kt*16+m, contraction c
        v16bf a;
        const bf16* ap = &sClus[(kt * 16 + m) * C_ + cb + 16 * g];
#pragma unroll
        for (int hh = 0; hh < 8; ++hh)
          ld_bf16x2(a, 2 * hh, ap + 2 * hh);
#pragma unroll
        for (int j = 0; j < 4; ++j) {
          int nt = 2 * j + par;
          // B fragment: x rows c = cb+16g+h (strided), col n = nt*16+m
          v16bf bb;
          const bf16* bp = &sX[(cb + 16 * g) * NTILE + nt * 16 + m];
#pragma unroll
          for (int h = 0; h < 16; ++h) bb[h] = bp[h * NTILE];
          acc[j] = __builtin_amdgcn_wmma_f32_16x16x32_bf16(
              false, a, false, bb, (short)0, acc[j], false, false);
        }
      }
      // write S tiles: D layout M = r + 8g, N = m
#pragma unroll
      for (int j = 0; j < 4; ++j) {
        int nt = 2 * j + par;
#pragma unroll
        for (int r = 0; r < 8; ++r)
          sS[(kt * 16 + 8 * g + r) * NTILE + nt * 16 + m] = acc[j][r];
      }
    }
    __syncthreads();

    // ---- phase 2: per-column softmax over K_ ----
    if (tid < NTILE) {
      int n = tid;
      float xsq = sXsq[n];
      float mx = -3.4e38f;
      for (int k = 0; k < K_; ++k) {
        float d = sScale[k] * (xsq - 2.f * sS[k * NTILE + n] + sCsq[k]);
        sS[k * NTILE + n] = d;
        mx = fmaxf(mx, d);
      }
      float sum = 0.f;
      for (int k = 0; k < K_; ++k) {
        float ev = __expf(sS[k * NTILE + n] - mx);
        sS[k * NTILE + n] = ev;
        sum += ev;
      }
      float inv = 1.f / sum;
      for (int k = 0; k < K_; ++k)
        sTheta[k * NTILE + n] = (bf16)(sS[k * NTILE + n] * inv);
    }
    __syncthreads();

    // chunk-local theta_sum accumulation
    if (tid < K_) {
      float s = 0.f;
      for (int n = 0; n < NTILE; ++n) s += (float)sTheta[tid * NTILE + n];
      sTsum[tid] += s;
    }

    // ---- phase 3: E(C_ x K_) += xtile(C_ x NTILE) * theta(NTILE x K_) ----
    for (int nb = 0; nb < NTILE; nb += 32) {
      // B fragment: theta rows n = nb+16g+h, col k = kt*16+m (contiguous in n)
      v16bf bb;
      const bf16* bp = &sTheta[(kt * 16 + m) * NTILE + nb + 16 * g];
#pragma unroll
      for (int hh = 0; hh < 8; ++hh) ld_bf16x2(bb, 2 * hh, bp + 2 * hh);
#pragma unroll
      for (int j = 0; j < 8; ++j) {
        int ct = 2 * j + par;
        // A fragment: x row c = ct*16+m, contraction n (contiguous pairs)
        v16bf a;
        const bf16* ap = &sX[(ct * 16 + m) * NTILE + nb + 16 * g];
#pragma unroll
        for (int hh = 0; hh < 8; ++hh)
          ld_bf16x2(a, 2 * hh, ap + 2 * hh);
        e[j] = __builtin_amdgcn_wmma_f32_16x16x32_bf16(
            false, a, false, bb, (short)0, e[j], false, false);
      }
    }
    __syncthreads();   // protect sX / sS / sTheta / sTsum for next tile
  }

  // ---- fold  -theta_sum_chunk ⊗ clusters  and atomically reduce to out ----
  {
    int k = kt * 16 + m;          // N dim of E tiles
    float ts = sTsum[k];
#pragma unroll
    for (int j = 0; j < 8; ++j) {
      int ct = 2 * j + par;
#pragma unroll
      for (int r = 0; r < 8; ++r) {
        int c = ct * 16 + 8 * g + r;   // M dim of E tiles
        float val = e[j][r] - ts * clusters[k * C_ + c];
        atomicAdd(&outb[k * C_ + c], val);
      }
    }
  }
}

extern "C" void kernel_launch(void* const* d_in, const int* in_sizes, int n_in,
                              void* d_out, int out_size, void* d_ws, size_t ws_size,
                              hipStream_t stream) {
  (void)in_sizes; (void)n_in; (void)d_ws; (void)ws_size;
  const float* x        = (const float*)d_in[0];
  const float* clusters = (const float*)d_in[1];
  const float* scale    = (const float*)d_in[2];
  float*       out      = (float*)d_out;

  // out is accumulated with atomics; zero it first (graph-capture safe)
  hipMemsetAsync(d_out, 0, (size_t)out_size * sizeof(float), stream);

  dim3 grid(B_ * NCHUNKS_PER_B);   // 32 batches x 8 N-chunks = 256 blocks
  dim3 block(256);                 // 8 waves
  surface_encoding_fused<<<grid, block, SMEM_BYTES, stream>>>(x, clusters, scale, out);
}
